// SingleVQC_8_12077448036872
// MI455X (gfx1250) — matrix operand order repaired and checked by reading, not compiled, not run on hardware
//
#include <hip/hip_runtime.h>

typedef __attribute__((ext_vector_type(16))) _Float16 v16h;
typedef __attribute__((ext_vector_type(8)))  float    v8f;

#define NQ     8
#define DIM    256
#define DEPTH  4
#define NITER  4
#define BATCH  32768
#define BM     256              // batch rows per block (2 M-tiles per wave)
#define NBLK   (BATCH / BM)     // 128 blocks

// halves per (iter,part) packed-U fragment buffer: 8 ksteps * 16 ntiles * 32 lanes * 16
#define FRAG_HALVES (8 * 16 * 32 * 16)      // 65536
#define FRAG_ITER   (2 * FRAG_HALVES)       // re + im per iteration

// MULTIPLIER = float32(pi - float32 eps)
#define MULT ((float)(3.14159265358979323846 - 1.1920928955078125e-07))

// ---------------------------------------------------------------------------
// Swizzle: (row m, col k in [0,256)) -> half-index in a tile stored as WMMA
// 16-bit A-fragments (16x32), fragment id = (mtile*8 + kstep), 32 lanes x 16
// halves contiguous per fragment.
// A layout (contiguous K, matches B): lane = mrow + 16*halfsel,
// halfsel = bit4 of k%32, element e = k%32 - 16*halfsel.
// ---------------------------------------------------------------------------
__device__ __forceinline__ int swizA(int m, int k) {
  int mtile = m >> 4, mrow = m & 15;
  int ks = k >> 5,   kl = k & 31;
  int hs = kl >> 4;
  int e  = kl & 15;
  int lane = mrow + 16 * hs;
  return (((mtile * 8 + ks) * 32 + lane) << 4) + e;
}

// ---------------------------------------------------------------------------
// Kernel 1: build U(theta_i)^T packed as WMMA B-fragments (f16), Re and Im.
// grid = (8 kgroups, 4 iters), 256 threads. Each block simulates 32 basis
// states (columns of U == rows of U^T) through the depth-4 circuit in LDS.
// ---------------------------------------------------------------------------
__global__ __launch_bounds__(256) void build_unitaries(
    const float* __restrict__ theta, _Float16* __restrict__ bfrags) {
  __shared__ float sre[32][DIM];
  __shared__ float sim_[32][DIM];
  const int tid = threadIdx.x;
  const int kgroup = blockIdx.x;            // 0..7 -> basis states 32*kgroup..
  const int iter = blockIdx.y;              // 0..3
  const float* w = theta + iter * (DEPTH * NQ * 3);

  for (int idx = tid; idx < 32 * DIM; idx += 256) {
    int col = idx >> 8, n = idx & 255;
    sre[col][n] = (n == (kgroup * 32 + col)) ? 1.0f : 0.0f;
    sim_[col][n] = 0.0f;
  }
  __syncthreads();

  for (int l = 0; l < DEPTH; ++l) {
    // Rot gates per qubit: g00=e_pp*ct, g01=-e_pm*st, g10=conj(e_pm)*st,
    // g11=conj(e_pp)*ct; e_pp=exp(-0.5i(phi+om)), e_pm=exp(0.5i(phi-om)).
    float g[NQ][8];
#pragma unroll
    for (int q = 0; q < NQ; ++q) {
      float phi = w[(l * NQ + q) * 3 + 0];
      float th  = w[(l * NQ + q) * 3 + 1];
      float om  = w[(l * NQ + q) * 3 + 2];
      float st, ct; sincosf(0.5f * th, &st, &ct);
      float sa, ca; sincosf(-0.5f * (phi + om), &sa, &ca);   // e_pp
      float sb, cb; sincosf( 0.5f * (phi - om), &sb, &cb);   // e_pm
      g[q][0] =  ca * ct; g[q][1] =  sa * ct;     // g00
      g[q][2] = -cb * st; g[q][3] = -sb * st;     // g01
      g[q][4] =  cb * st; g[q][5] = -sb * st;     // g10
      g[q][6] =  ca * ct; g[q][7] = -sa * ct;     // g11
    }
#pragma unroll
    for (int q = 0; q < NQ; ++q) {                // apply 2x2 complex gates
      const int stride = 1 << (7 - q);
      for (int widx = tid; widx < 32 * 128; widx += 256) {
        int col = widx >> 7, p = widx & 127;
        int a = p >> (7 - q), c2 = p & (stride - 1);
        int i0 = a * (stride << 1) + c2, i1 = i0 + stride;
        float v0r = sre[col][i0], v0i = sim_[col][i0];
        float v1r = sre[col][i1], v1i = sim_[col][i1];
        sre[col][i0] = g[q][0]*v0r - g[q][1]*v0i + g[q][2]*v1r - g[q][3]*v1i;
        sim_[col][i0] = g[q][0]*v0i + g[q][1]*v0r + g[q][2]*v1i + g[q][3]*v1r;
        sre[col][i1] = g[q][4]*v0r - g[q][5]*v0i + g[q][6]*v1r - g[q][7]*v1i;
        sim_[col][i1] = g[q][4]*v0i + g[q][5]*v0r + g[q][6]*v1i + g[q][7]*v1r;
      }
      __syncthreads();
    }
    // CNOT ring, control q -> target (q+r)%8, r = l%7+1, applied sequentially
    const int r = (l % 7) + 1;
    for (int q = 0; q < NQ; ++q) {
      const int pc = 7 - q, pt = 7 - ((q + r) & 7);   // bit positions (q0=MSB)
      for (int widx = tid; widx < 32 * 64; widx += 256) {
        int col = widx >> 6, p = widx & 63;
        int n = (1 << pc), pp = p;
        for (int b = 0; b < 8; ++b) {
          if (b == pc || b == pt) continue;
          n |= (pp & 1) << b; pp >>= 1;
        }
        int n2 = n | (1 << pt);                       // swap (c=1,t=0)<->(c=1,t=1)
        float tr = sre[col][n], ti = sim_[col][n];
        sre[col][n] = sre[col][n2];  sim_[col][n] = sim_[col][n2];
        sre[col][n2] = tr;           sim_[col][n2] = ti;
      }
      __syncthreads();
    }
  }

  // Pack B-fragments: B[k][n] = U[n][k]; simulated column k gives row k of B.
  // B packed in contiguous K order (as swizA): lane<16 -> K = e, lane>=16 -> K = e+16; n = lane&15.
  _Float16* out = bfrags + (size_t)iter * FRAG_ITER;
  for (int idx = tid; idx < 2 * 16 * 32 * 16; idx += 256) {
    int part  = idx >> 13;
    int rem   = idx & 8191;
    int ntile = rem >> 9;
    int lane  = (rem >> 4) & 31;
    int e     = rem & 15;
    int col   = e + 16 * (lane >> 4);                 // k within this kgroup
    int n     = ntile * 16 + (lane & 15);
    float val = part ? sim_[col][n] : sre[col][n];
    out[(size_t)part * FRAG_HALVES +
        (((kgroup * 16 + ntile) * 32 + lane) << 4) + e] = (_Float16)val;
  }
}

// ---------------------------------------------------------------------------
// Kernel 2: one VQC iteration for the whole batch.
//   s0 (real product state) -> s1 = s0 @ U^T via WMMA (Re,Im) -> p=|s1|^2
//   -> expvals = p @ Zsign via WMMA -> xnext (or final scaled output).
// 256 threads = 8 waves; wave w owns TWO M-tiles (rows [32w,32w+32)) so each
// B-fragment load feeds 4 WMMAs -> halves L2 traffic vs 1 tile/wave.
// All LDS use is wave-local; 256 KB LDS/block (within 320 KB/WGP).
// ---------------------------------------------------------------------------
__global__ __launch_bounds__(256) void vqc_step(
    const float* __restrict__ x, float* __restrict__ xnext,
    float* __restrict__ dout, const _Float16* __restrict__ bfrags, int last) {
  __shared__ _Float16 sA[BM * DIM];   // 128 KB, swizzled A-fragments (s0)
  __shared__ _Float16 sP[BM * DIM];   // 128 KB, swizzled A-fragments (p)
  const int tid = threadIdx.x;
  const int lane = tid & 31;
  const int wave = tid >> 5;

  // ---- phase 0: product-state amplitudes (real) into swizzled A tile ----
  {
    const int m = tid;                      // one local row per thread
    const int b = blockIdx.x * BM + m;
    float c[NQ], s[NQ];
#pragma unroll
    for (int q = 0; q < NQ; ++q) sincosf(0.5f * x[b * NQ + q], &s[q], &c[q]);
    float lo[16], hi[16];
#pragma unroll
    for (int j = 0; j < 16; ++j) {
      lo[j] = (((j >> 3) & 1) ? s[4] : c[4]) * (((j >> 2) & 1) ? s[5] : c[5]) *
              (((j >> 1) & 1) ? s[6] : c[6]) * ((j & 1) ? s[7] : c[7]);
      hi[j] = (((j >> 3) & 1) ? s[0] : c[0]) * (((j >> 2) & 1) ? s[1] : c[1]) *
              (((j >> 1) & 1) ? s[2] : c[2]) * ((j & 1) ? s[3] : c[3]);
    }
#pragma unroll
    for (int i = 0; i < 16; ++i)
#pragma unroll
      for (int j = 0; j < 16; ++j)
        sA[swizA(m, i * 16 + j)] = (_Float16)(hi[i] * lo[j]);
  }
  __syncthreads();

  // ---- phase 1: s1 = s0 @ U^T (Re & Im GEMMs), p = |s1|^2 into sP ----
  const _Float16* bre = bfrags;
  const _Float16* bim = bfrags + FRAG_HALVES;
  const v16h* sAv = (const v16h*)sA;
  v16h afrag[2][8];                                  // two M-tiles per wave
#pragma unroll
  for (int mt = 0; mt < 2; ++mt)
#pragma unroll
    for (int ks = 0; ks < 8; ++ks)
      afrag[mt][ks] = sAv[((wave * 2 + mt) * 8 + ks) * 32 + lane];

  for (int ntile = 0; ntile < 16; ++ntile) {
    v8f accr[2] = {{}, {}};
    v8f acci[2] = {{}, {}};
#pragma unroll
    for (int ks = 0; ks < 8; ++ks) {
      v16h bfr = *(const v16h*)(bre + ((((ks * 16 + ntile) * 32) + lane) << 4));
      v16h bfi = *(const v16h*)(bim + ((((ks * 16 + ntile) * 32) + lane) << 4));
#pragma unroll
      for (int mt = 0; mt < 2; ++mt) {
        accr[mt] = __builtin_amdgcn_wmma_f32_16x16x32_f16(
            false, afrag[mt][ks], false, bfr, (short)0, accr[mt], false, false);
        acci[mt] = __builtin_amdgcn_wmma_f32_16x16x32_f16(
            false, afrag[mt][ks], false, bfi, (short)0, acci[mt], false, false);
      }
    }
#pragma unroll
    for (int mt = 0; mt < 2; ++mt)
#pragma unroll
      for (int v = 0; v < 8; ++v) {
        float re = accr[mt][v], im = acci[mt][v];
        int row  = (wave * 2 + mt) * 16 + v + 8 * (lane >> 4);
        int coln = ntile * 16 + (lane & 15);
        sP[swizA(row, coln)] = (_Float16)(re * re + im * im);
      }
  }
  __syncthreads();

  // ---- phase 2: expvals = p @ Zsign (Zsign built inline, f32 accumulate) ---
  {
    v8f acc[2] = {{}, {}};
    const v16h* sPv = (const v16h*)sP;
    const int nq = lane & 15;
#pragma unroll
    for (int ks = 0; ks < 8; ++ks) {
      v16h zf;
#pragma unroll
      for (int e = 0; e < 16; ++e) {
        int k = ks * 32 + e + 16 * (lane >> 4);       // B layout K index
        float zv = (nq < NQ) ? (((k >> (7 - nq)) & 1) ? -1.f : 1.f) : 0.f;
        zf[e] = (_Float16)zv;
      }
#pragma unroll
      for (int mt = 0; mt < 2; ++mt) {
        v16h pf = sPv[((wave * 2 + mt) * 8 + ks) * 32 + lane];
        acc[mt] = __builtin_amdgcn_wmma_f32_16x16x32_f16(
            false, pf, false, zf, (short)0, acc[mt], false, false);
      }
    }
#pragma unroll
    for (int mt = 0; mt < 2; ++mt)
#pragma unroll
      for (int v = 0; v < 8; ++v) {
        int row = (wave * 2 + mt) * 16 + v + 8 * (lane >> 4);
        int b = blockIdx.x * BM + row;
        if (!last) {
          if (nq < NQ) xnext[b * NQ + nq] = acc[mt][v];
        } else {
          if (nq == 4) dout[b] = acc[mt][v] * MULT;
        }
      }
  }
}

// ---------------------------------------------------------------------------
extern "C" void kernel_launch(void* const* d_in, const int* in_sizes, int n_in,
                              void* d_out, int out_size, void* d_ws, size_t ws_size,
                              hipStream_t stream) {
  const float* x0    = (const float*)d_in[0];   // (32768, 8) f32
  const float* theta = (const float*)d_in[1];   // (4, 4, 8, 3) f32
  _Float16* bfrags = (_Float16*)d_ws;                               // 1 MB
  float* xb0 = (float*)((char*)d_ws + (size_t)NITER * FRAG_ITER * sizeof(_Float16));
  float* xb1 = xb0 + (size_t)BATCH * NQ;                            // 2 x 1 MB

  build_unitaries<<<dim3(8, NITER), 256, 0, stream>>>(theta, bfrags);
  vqc_step<<<NBLK, 256, 0, stream>>>(x0,  xb0, (float*)d_out, bfrags + 0 * (size_t)FRAG_ITER, 0);
  vqc_step<<<NBLK, 256, 0, stream>>>(xb0, xb1, (float*)d_out, bfrags + 1 * (size_t)FRAG_ITER, 0);
  vqc_step<<<NBLK, 256, 0, stream>>>(xb1, xb0, (float*)d_out, bfrags + 2 * (size_t)FRAG_ITER, 0);
  vqc_step<<<NBLK, 256, 0, stream>>>(xb0, xb1, (float*)d_out, bfrags + 3 * (size_t)FRAG_ITER, 1);
}
